// EncdecMultiheadAttn_78546361909874
// MI455X (gfx1250) — matrix operand order repaired and checked by reading, not compiled, not run on hardware
//
#include <hip/hip_runtime.h>
#include <math.h>

// Problem constants (fixed by the reference).
#define LQ_  1024
#define LK_  1024
#define BB_  4
#define EE_  1024
#define HH_  16
#define DH_  64

// ---------------------------------------------------------------------------
// CDNA5 WMMA types: wave32, 16x16x32 bf16 -> f32 accum.
//   A frag: 16 bf16 / lane (8 VGPRs). Lane holds row m = lane%16,
//           k = 16 contiguous starting at (lane&16)?16:0 (this code's order, not AMD's table).
//   B frag: 16 bf16 / lane. Lane holds col n = lane%16,
//           k = 16 contiguous starting at (lane&16)?16:0 (same order as A; any shared order works).
//   C/D:    8 f32 / lane. c[r] = (M = r + 8*(lane/16), N = lane%16).
// ---------------------------------------------------------------------------
typedef __attribute__((ext_vector_type(16))) __bf16 v16bf;
typedef __attribute__((ext_vector_type(8)))  __bf16 bf16x8;
typedef __attribute__((ext_vector_type(8)))  float  v8f;
typedef __attribute__((ext_vector_type(4)))  float  f32x4;

static __device__ __forceinline__ v8f wmma_bf16(v16bf a, v16bf b, v8f c) {
  return __builtin_amdgcn_wmma_f32_16x16x32_bf16(
      /*neg_a=*/false, a, /*neg_b=*/false, b,
      /*c_mod=*/(short)0, c, /*reuse_a=*/false, /*reuse_b=*/false);
}

static __device__ __forceinline__ v16bf combine16(bf16x8 lo, bf16x8 hi) {
  return __builtin_shufflevector(lo, hi, 0,1,2,3,4,5,6,7,8,9,10,11,12,13,14,15);
}

// 8 contiguous fp32 -> 8 bf16 (two 16B vector loads, RNE converts).
static __device__ __forceinline__ bf16x8 cvt8(const float* p) {
  f32x4 a = *(const f32x4*)p;
  f32x4 b = *(const f32x4*)(p + 4);
  bf16x8 r;
  r[0]=(__bf16)a[0]; r[1]=(__bf16)a[1]; r[2]=(__bf16)a[2]; r[3]=(__bf16)a[3];
  r[4]=(__bf16)b[0]; r[5]=(__bf16)b[1]; r[6]=(__bf16)b[2]; r[7]=(__bf16)b[3];
  return r;
}

// A fragment from fp32 row-major source (rowPtr = &A[m_of_lane][0]).
static __device__ __forceinline__ v16bf load_a_f32(const float* rowPtr, int k0, int lane) {
  const int koff = k0 + ((lane & 16) ? 16 : 0);
  return combine16(cvt8(rowPtr + koff), cvt8(rowPtr + koff + 8));
}

// B fragment from fp32 row-major "weight" matrix W[n][k] (NT gemm: k contiguous).
static __device__ __forceinline__ v16bf load_b_f32(const float* W, int ld, int n0, int k0, int lane) {
  const float* p = W + (size_t)(n0 + (lane & 15)) * ld + k0 + ((lane & 16) ? 16 : 0);
  return combine16(cvt8(p), cvt8(p + 8));
}

// A fragment from bf16 row-major source.
static __device__ __forceinline__ v16bf load_a_bf16(const __bf16* rowPtr, int k0, int lane) {
  const int koff = k0 + ((lane & 16) ? 16 : 0);
  return combine16(*(const bf16x8*)(rowPtr + koff), *(const bf16x8*)(rowPtr + koff + 8));
}

// B fragment from bf16 row-major source (rows = n, k contiguous).
static __device__ __forceinline__ v16bf load_b_bf16(const __bf16* base, int ld, int n0, int k0, int lane) {
  const __bf16* p = base + (size_t)(n0 + (lane & 15)) * ld + k0 + ((lane & 16) ? 16 : 0);
  return combine16(*(const bf16x8*)p, *(const bf16x8*)(p + 8));
}

// ---------------------------------------------------------------------------
// Projection GEMM: C[M,N] = A[M,EE_] * W[N,EE_]^T, bf16 WMMA, fp32 accum.
// Block = 128 threads (4 waves); wave tile = 16 rows x 64 cols.
// L2 (192MB) holds every operand, so no LDS staging needed to hit roofline.
// MODE 0: scatter Q  -> bf16 [b,h,LQ,DH]
// MODE 1: scatter KV -> K bf16 [b,h,LK,DH], V bf16 transposed [b,h,DH,LK]
// MODE 2: out fp32 [M,EE_] (final projection)
// ---------------------------------------------------------------------------
template <int MODE>
__global__ __launch_bounds__(128) void proj_gemm_kernel(
    const float* __restrict__ A, const float* __restrict__ W,
    __bf16* __restrict__ dst0, __bf16* __restrict__ dst1,
    float* __restrict__ dstF)
{
  const int lane = threadIdx.x & 31;
  const int wave = threadIdx.x >> 5;
  const int m0 = blockIdx.y * 64 + wave * 16;
  const int n0 = blockIdx.x * 64;

  const float* Arow = A + (size_t)(m0 + (lane & 15)) * EE_;

  v8f acc0 = {0,0,0,0,0,0,0,0};
  v8f acc1 = {0,0,0,0,0,0,0,0};
  v8f acc2 = {0,0,0,0,0,0,0,0};
  v8f acc3 = {0,0,0,0,0,0,0,0};

  for (int k = 0; k < EE_; k += 32) {
    v16bf a  = load_a_f32(Arow, k, lane);
    v16bf b0 = load_b_f32(W, EE_, n0 +  0, k, lane);
    v16bf b1 = load_b_f32(W, EE_, n0 + 16, k, lane);
    v16bf b2 = load_b_f32(W, EE_, n0 + 32, k, lane);
    v16bf b3 = load_b_f32(W, EE_, n0 + 48, k, lane);
    acc0 = wmma_bf16(a, b0, acc0);
    acc1 = wmma_bf16(a, b1, acc1);
    acc2 = wmma_bf16(a, b2, acc2);
    acc3 = wmma_bf16(a, b3, acc3);
  }

  const int mhi = m0 + ((lane & 16) ? 8 : 0);
  #pragma unroll
  for (int t = 0; t < 4; ++t) {
    v8f acc = (t == 0) ? acc0 : (t == 1) ? acc1 : (t == 2) ? acc2 : acc3;
    const int n = n0 + t * 16 + (lane & 15);
    #pragma unroll
    for (int r = 0; r < 8; ++r) {
      const int m = mhi + r;           // m = q*B + b (row-major (q,b) flatten)
      const float val = acc[r];
      if constexpr (MODE == 0) {
        const int q = m >> 2, b = m & 3;
        const int head = n >> 6, d = n & 63;
        dst0[(((size_t)(b * HH_ + head)) * LQ_ + q) * DH_ + d] = (__bf16)val;
      } else if constexpr (MODE == 1) {
        const int kk = m >> 2, b = m & 3;
        const int head = n >> 7, which = (n >> 6) & 1, d = n & 63;
        if (which == 0)
          dst0[(((size_t)(b * HH_ + head)) * LK_ + kk) * DH_ + d] = (__bf16)val;
        else
          dst1[(((size_t)(b * HH_ + head)) * DH_ + d) * LK_ + kk] = (__bf16)val;
      } else {
        dstF[(size_t)m * EE_ + n] = val;
      }
    }
  }
}

// ---------------------------------------------------------------------------
// Fused attention: one block (4 waves, 128 thr) per (16 q-rows, b*H+h).
//  P1: S = Q K^T (bf16 WMMA, K=DH=64) -> padded LDS (bank-conflict free)
//  P2: scale+mask+softmax in LDS, coalesced fp32 coverage write (the 256MB
//      HBM-bound part of this problem)
//  P3: ctx = P V via bf16 WMMA against pre-transposed V
// ---------------------------------------------------------------------------
#define SPAD 4   // LDS row pad: stride % 64banks = 4 -> conflict-free

__global__ __launch_bounds__(128) void attn_softmax_kernel(
    const __bf16* __restrict__ Qb, const __bf16* __restrict__ Kb,
    const __bf16* __restrict__ Vtb, const unsigned char* __restrict__ mask,
    float* __restrict__ ctx, float* __restrict__ coverage)
{
  __shared__ float S[16][LK_ + SPAD];   // scores -> probs (fp32)
  __shared__ float red[16][8];
  __shared__ float rowstat[16];

  const int tid  = threadIdx.x;
  const int lane = tid & 31;
  const int wave = tid >> 5;
  const int qt   = blockIdx.x;       // 16-row q tile
  const int bh   = blockIdx.y;       // b*H + h
  const int b    = bh >> 4;
  const int h    = bh & 15;
  const int mhi  = (lane & 16) ? 8 : 0;

  // ---- Phase 1: S = Q K^T ----
  const __bf16* Qrow  = Qb + ((size_t)bh * LQ_ + qt * 16 + (lane & 15)) * DH_;
  const v16bf qa0 = load_a_bf16(Qrow, 0, lane);
  const v16bf qa1 = load_a_bf16(Qrow, 32, lane);
  const __bf16* Kbase = Kb + (size_t)bh * LK_ * DH_;

  for (int t = 0; t < 16; ++t) {
    const int n0 = wave * 256 + t * 16;
    v16bf bk0 = load_b_bf16(Kbase, DH_, n0, 0, lane);
    v16bf bk1 = load_b_bf16(Kbase, DH_, n0, 32, lane);
    v8f c = {0,0,0,0,0,0,0,0};
    c = wmma_bf16(qa0, bk0, c);
    c = wmma_bf16(qa1, bk1, c);
    const int col = n0 + (lane & 15);
    #pragma unroll
    for (int r = 0; r < 8; ++r) S[mhi + r][col] = c[r];
  }
  __syncthreads();

  // ---- Phase 2: softmax over LK per row (8 threads / row) ----
  const float scale = 0.125f;                 // 1/sqrt(DH)
  const int row = tid >> 3;
  const int sub = tid & 7;
  const unsigned char* mrow = mask + b * LK_;

  float mx = -__builtin_inff();
  for (int j = 0; j < 128; ++j) {
    const int c = sub * 128 + j;
    float s = S[row][c] * scale;
    if (mrow[c]) s = -__builtin_inff();
    S[row][c] = s;
    mx = fmaxf(mx, s);
  }
  red[row][sub] = mx;
  __syncthreads();
  if (sub == 0) {
    float m = red[row][0];
    #pragma unroll
    for (int i = 1; i < 8; ++i) m = fmaxf(m, red[row][i]);
    rowstat[row] = m;
  }
  __syncthreads();
  const float rm = rowstat[row];
  float sum = 0.f;
  for (int j = 0; j < 128; ++j) {
    const int c = sub * 128 + j;
    const float p = __expf(S[row][c] - rm);
    S[row][c] = p;
    sum += p;
  }
  red[row][sub] = sum;
  __syncthreads();
  if (sub == 0) {
    float s = 0.f;
    #pragma unroll
    for (int i = 0; i < 8; ++i) s += red[row][i];
    rowstat[row] = 1.0f / s;
  }
  __syncthreads();

  // ---- Phase 2b: normalize + coalesced coverage store ----
  float* cov = coverage + ((size_t)bh * LQ_ + qt * 16) * LK_;
  for (int idx = tid; idx < 16 * LK_; idx += 128) {
    const int r = idx >> 10;
    const int c = idx & (LK_ - 1);
    const float p = S[r][c] * rowstat[r];
    S[r][c] = p;
    cov[(size_t)r * LK_ + c] = p;
  }
  __syncthreads();

  // ---- Phase 3: ctx = P V (each wave owns a 16-wide d slice) ----
  const __bf16* Vbase = Vtb + (size_t)bh * DH_ * LK_;   // [DH][LK]
  const int d0 = wave * 16;
  const float* Srow = &S[lane & 15][0];
  v8f acc = {0,0,0,0,0,0,0,0};
  for (int kk = 0; kk < LK_; kk += 32) {
    v16bf a  = load_a_f32(Srow, kk, lane);              // LDS fp32 -> bf16 frag
    v16bf bv = load_b_bf16(Vbase, LK_, d0, kk, lane);
    acc = wmma_bf16(a, bv, acc);
  }
  const int n = d0 + (lane & 15);
  #pragma unroll
  for (int r = 0; r < 8; ++r) {
    const int q = qt * 16 + mhi + r;
    ctx[((size_t)q * BB_ + b) * EE_ + h * DH_ + n] = acc[r];
  }
}

// ---------------------------------------------------------------------------
// Launch. ws layout (needs 40 MiB):
//   [0,8Mi)   Qb  bf16 [b,h,LQ,DH]
//   [8,16Mi)  Kb  bf16 [b,h,LK,DH]
//   [16,24Mi) Vtb bf16 [b,h,DH,LK]
//   [24,40Mi) ctx fp32 [LQ*B, E]
// d_out = out fp32 (LQ*B*E) ++ coverage fp32 (B*H*LQ*LK)
// ---------------------------------------------------------------------------
extern "C" void kernel_launch(void* const* d_in, const int* in_sizes, int n_in,
                              void* d_out, int out_size, void* d_ws, size_t ws_size,
                              hipStream_t stream) {
  const float* query = (const float*)d_in[0];
  const float* key   = (const float*)d_in[1];
  const float* Wq    = (const float*)d_in[2];
  const float* Wkv   = (const float*)d_in[3];
  const float* Wout  = (const float*)d_in[4];
  const unsigned char* mask = (const unsigned char*)d_in[5];
  // d_in[6] = num_heads (compile-time H = 16)

  float* out      = (float*)d_out;
  float* coverage = out + (size_t)LQ_ * BB_ * EE_;

  const size_t QKELEMS = (size_t)BB_ * HH_ * LQ_ * DH_;   // 4Mi elements
  __bf16* Qb  = (__bf16*)d_ws;
  __bf16* Kb  = Qb + QKELEMS;
  __bf16* Vtb = Kb + QKELEMS;
  float*  ctx = (float*)(Vtb + QKELEMS);

  // Q projection: (4096 x 1024) = query * Wq^T
  proj_gemm_kernel<0><<<dim3(EE_ / 64, (LQ_ * BB_) / 64), 128, 0, stream>>>(
      query, Wq, Qb, nullptr, nullptr);
  // KV projection: (4096 x 2048) = key * Wkv^T, split into K and V^T
  proj_gemm_kernel<1><<<dim3((2 * EE_) / 64, (LK_ * BB_) / 64), 128, 0, stream>>>(
      key, Wkv, Kb, Vtb, nullptr);
  // Fused scores + softmax + coverage + context
  attn_softmax_kernel<<<dim3(LQ_ / 16, BB_ * HH_), 128, 0, stream>>>(
      Qb, Kb, Vtb, mask, ctx, coverage);
  // Output projection: out = ctx * Wout^T
  proj_gemm_kernel<2><<<dim3(EE_ / 64, (LQ_ * BB_) / 64), 128, 0, stream>>>(
      ctx, Wout, nullptr, nullptr, out);
}
